// Seq2SeqAttentionDecoder_58686433132643
// MI455X (gfx1250) — matrix operand order repaired and checked by reading, not compiled, not run on hardware
//
#include <hip/hip_runtime.h>
#include <cstddef>

// ---------------------------------------------------------------------------
// Seq2Seq attention decoder (2-layer GRU + additive attention + vocab proj)
// CDNA5 (gfx1250) implementation: all GEMMs use V_WMMA_F32_16X16X32_BF16,
// weights converted once to bf16 (f32 accumulate), output GEMM is LDS-tiled
// with double buffering. Sequential T=64 recurrence handled by per-step
// kernel launches on `stream` (graph-capture friendly, no sync calls).
// ---------------------------------------------------------------------------

#define BSZ   32
#define TDEC  64
#define SENC  128
#define HDIM  1024
#define EDIM  512
#define VOCAB 32000
#define G3    (3 * HDIM)
#define HE    (HDIM + EDIM)
#define NEGV  (-1000000.0f)

typedef __bf16 bf16_t;
typedef __attribute__((ext_vector_type(16))) __bf16 v16bf;
typedef __attribute__((ext_vector_type(8)))  __bf16 v8bf;
typedef __attribute__((ext_vector_type(8)))  float  v8f;

// round-to-nearest-even f32 -> bf16 (bit pattern in u16)
__device__ __forceinline__ unsigned short f32_to_bf16(float f) {
    unsigned int u = __float_as_uint(f);
    unsigned int lsb = (u >> 16) & 1u;
    u += 0x7fffu + lsb;
    return (unsigned short)(u >> 16);
}

// A-fragment (16x32 MxK bf16): lane M = lane&15, half = lane>>4.
// frag elem i: K = half*8 + i (i<8), K = 16 + half*8 + (i-8) (i>=8).
__device__ __forceinline__ v16bf ldfragA(const bf16_t* p, int half) {
    v8bf lo = *(const v8bf*)(p + half * 8);
    v8bf hi = *(const v8bf*)(p + 16 + half * 8);
    v16bf r;
#pragma unroll
    for (int i = 0; i < 8; ++i) { r[i] = lo[i]; r[i + 8] = hi[i]; }
    return r;
}

// B-fragment (32x16 KxN bf16): lane N = lane&15, half = lane>>4.
// frag elem i: K = half*8 + i (i<8), K = 16 + half*8 + (i-8) (i>=8).
__device__ __forceinline__ v16bf ldfragB(const bf16_t* p, int half) {
    v8bf lo = *(const v8bf*)(p + half * 8);
    v8bf hi = *(const v8bf*)(p + 16 + half * 8);
    v16bf r;
#pragma unroll
    for (int i = 0; i < 8; ++i) { r[i] = lo[i]; r[i + 8] = hi[i]; }
    return r;
}

// ---------------------------------------------------------------------------
// elementwise f32 -> bf16
__global__ void s2s_cvt_bf16(const float* __restrict__ src,
                             unsigned short* __restrict__ dst, int n) {
    int i = blockIdx.x * 256 + threadIdx.x;
    if (i < n) dst[i] = f32_to_bf16(src[i]);
}

// embedding gather: xs[t][b][e] = bf16(emb[X[b][t]][e])
__global__ void s2s_embed(const int* __restrict__ X, const float* __restrict__ emb,
                          unsigned short* __restrict__ xs) {
    int idx = blockIdx.x * 256 + threadIdx.x;        // < T*B*E
    int e = idx & (EDIM - 1);
    int tb = idx >> 9;                                // t*B + b
    int b = tb & (BSZ - 1);
    int t = tb >> 5;
    int tok = X[b * TDEC + t];
    xs[idx] = f32_to_bf16(emb[(size_t)tok * EDIM + e]);
}

// hidden_state [2,B,H] f32 -> h0/h1 state (f32 + bf16 copies)
__global__ void s2s_init_hidden(const float* __restrict__ hid,
                                float* __restrict__ h0, float* __restrict__ h1,
                                unsigned short* __restrict__ h0b,
                                unsigned short* __restrict__ h1b) {
    int idx = blockIdx.x * 256 + threadIdx.x;        // < B*H
    float a = hid[idx];
    float c = hid[BSZ * HDIM + idx];
    h0[idx] = a; h1[idx] = c;
    h0b[idx] = f32_to_bf16(a); h1b[idx] = f32_to_bf16(c);
}

// ---------------------------------------------------------------------------
// Generic C[M,N] = A[M,K] * W[N,K]^T (+bias), bf16 in / f32 out, WMMA.
// Block = 256 threads = 8 waves tiled 2(M) x 4(N); M%32==0, N%64==0, K%32==0.
__global__ __launch_bounds__(256)
void s2s_gemm_bt(const unsigned short* __restrict__ Au,
                 const unsigned short* __restrict__ Wu,
                 const float* __restrict__ bias, float* __restrict__ C,
                 int M, int N, int K) {
    const bf16_t* A = (const bf16_t*)Au;
    const bf16_t* W = (const bf16_t*)Wu;
    int w = threadIdx.x >> 5, lane = threadIdx.x & 31;
    int half = lane >> 4, mr = lane & 15;
    int row0 = blockIdx.y * 32 + (w >> 2) * 16;
    int col0 = blockIdx.x * 64 + (w & 3) * 16;
    const bf16_t* arow = A + (size_t)(row0 + mr) * K;
    const bf16_t* brow = W + (size_t)(col0 + mr) * K;
    v8f acc = {};
    for (int k0 = 0; k0 < K; k0 += 32) {
        if (k0 + 128 < K) {          // WGP-scope prefetch -> all cache levels
            __builtin_prefetch(arow + k0 + 128, 0, 3);
            __builtin_prefetch(brow + k0 + 128, 0, 3);
        }
        v16bf a = ldfragA(arow + k0, half);
        v16bf b = ldfragB(brow + k0, half);
        acc = __builtin_amdgcn_wmma_f32_16x16x32_bf16(
            false, a, false, b, (short)0, acc, false, false);
    }
    int col = col0 + mr;
    float bv = bias ? bias[col] : 0.0f;
#pragma unroll
    for (int r = 0; r < 8; ++r) {
        int row = row0 + half * 8 + r;
        C[(size_t)row * N + col] = acc[r] + bv;
    }
}

// ---------------------------------------------------------------------------
// Attention: scores = wv . tanh(q + keys), masked softmax, ctx = attn @ enc.
// One block per batch row; 8 waves, wave32 shuffle reduction over H.
__global__ __launch_bounds__(256)
void s2s_attn(const float* __restrict__ q, const float* __restrict__ keys,
              const float* __restrict__ enc, const float* __restrict__ wv,
              const int* __restrict__ valid_lens, float* __restrict__ ctx) {
    __shared__ float sc[SENC];
    int b = blockIdx.x;
    int tid = threadIdx.x, w = tid >> 5, lane = tid & 31;
    const float* qb = q + (size_t)b * HDIM;
    const float* kb = keys + (size_t)b * SENC * HDIM;
    for (int s = w; s < SENC; s += 8) {
        const float* ks = kb + (size_t)s * HDIM;
        float p = 0.0f;
        for (int h = lane; h < HDIM; h += 32)
            p += tanhf(qb[h] + ks[h]) * wv[h];
#pragma unroll
        for (int off = 16; off > 0; off >>= 1)
            p += __shfl_down(p, off, 32);
        if (lane == 0) sc[s] = p;
    }
    __syncthreads();
    int vl = valid_lens[b];
    float mx = -3.0e38f;
    for (int s = 0; s < SENC; ++s) {
        float v = (s < vl) ? sc[s] : NEGV;
        mx = fmaxf(mx, v);
    }
    float ssum = 0.0f;
    for (int s = 0; s < SENC; ++s) {
        float v = (s < vl) ? sc[s] : NEGV;
        ssum += __expf(v - mx);
    }
    float inv = 1.0f / ssum;
    const float* eb = enc + (size_t)b * SENC * HDIM;
    for (int h = tid; h < HDIM; h += 256) {
        float acc = 0.0f;
        for (int s = 0; s < SENC; ++s) {
            float v = (s < vl) ? sc[s] : NEGV;
            acc += __expf(v - mx) * eb[(size_t)s * HDIM + h];
        }
        ctx[(size_t)b * HDIM + h] = acc * inv;
    }
}

// pack GRU layer-0 input: inp[b][0:H]=bf16(ctx), inp[b][H:H+E]=xs[t][b]
__global__ void s2s_pack_inp(const float* __restrict__ ctx,
                             const unsigned short* __restrict__ xs_t,
                             unsigned short* __restrict__ inp) {
    int idx = blockIdx.x * 256 + threadIdx.x;        // < B*(H+E)
    int b = idx / HE, c = idx - b * HE;
    unsigned short v = (c < HDIM) ? f32_to_bf16(ctx[b * HDIM + c])
                                  : xs_t[b * EDIM + (c - HDIM)];
    inp[idx] = v;
}

// fused GRU gate math (PyTorch convention); writes f32 state + bf16 copy
__global__ void s2s_gru_update(const float* __restrict__ gi,
                               const float* __restrict__ gh,
                               const float* __restrict__ hprev,
                               float* __restrict__ hnew,
                               unsigned short* __restrict__ hnew_b,
                               unsigned short* __restrict__ out_b) {
    int idx = blockIdx.x * 256 + threadIdx.x;        // < B*H
    int b = idx >> 10, hh = idx & (HDIM - 1);
    const float* gib = gi + (size_t)b * G3;
    const float* ghb = gh + (size_t)b * G3;
    float ir = gib[hh], iz = gib[HDIM + hh], inn = gib[2 * HDIM + hh];
    float hr = ghb[hh], hz = ghb[HDIM + hh], hn = ghb[2 * HDIM + hh];
    float r = 1.0f / (1.0f + __expf(-(ir + hr)));
    float z = 1.0f / (1.0f + __expf(-(iz + hz)));
    float n = tanhf(inn + r * hn);
    float h = (1.0f - z) * n + z * hprev[idx];
    hnew[idx] = h;
    unsigned short hb = f32_to_bf16(h);
    hnew_b[idx] = hb;
    if (out_b) out_b[idx] = hb;
}

// ---------------------------------------------------------------------------
// Final projection: out[b][t][v] = outs[t*32+b] . W_out[v] + b_out[v]
// LDS-tiled (64x128x32, double buffered), 8 waves x 2x2 WMMA frags each.
#define OBM 64
#define OBN 128
#define OBK 32
__global__ __launch_bounds__(256)
void s2s_out_gemm(const unsigned short* __restrict__ Au,
                  const unsigned short* __restrict__ Wu,
                  const float* __restrict__ bias, float* __restrict__ out,
                  int K) {
    __shared__ unsigned short sA[2][OBM * OBK];
    __shared__ unsigned short sB[2][OBN * OBK];
    const bf16_t* A = (const bf16_t*)Au;
    const bf16_t* W = (const bf16_t*)Wu;
    int tid = threadIdx.x;
    int bm = blockIdx.y * OBM, bn = blockIdx.x * OBN;
    int w = tid >> 5, lane = tid & 31, half = lane >> 4, mr = lane & 15;
    int wm = (w >> 2) * 32, wn = (w & 3) * 32;
    v8f acc[2][2] = {};

    auto stage = [&](int buf, int k0) {
        {   // A tile: 64 rows x 4 chunks of 8 bf16 -> 256 chunks
            int row = tid >> 2, ch = tid & 3;
            const bf16_t* src = A + (size_t)(bm + row) * K + k0 + ch * 8;
            if (k0 + 2 * OBK < K) __builtin_prefetch(src + 2 * OBK, 0, 3);
            *(v8bf*)&sA[buf][row * OBK + ch * 8] = *(const v8bf*)src;
        }
#pragma unroll
        for (int i = 0; i < 2; ++i) {   // B tile: 128 rows x 4 chunks
            int c = tid + i * 256;
            int row = c >> 2, ch = c & 3;
            const bf16_t* src = W + (size_t)(bn + row) * K + k0 + ch * 8;
            if (k0 + 2 * OBK < K) __builtin_prefetch(src + 2 * OBK, 0, 3);
            *(v8bf*)&sB[buf][row * OBK + ch * 8] = *(const v8bf*)src;
        }
    };

    stage(0, 0);
    __syncthreads();
    int KT = K / OBK;
    for (int kt = 0; kt < KT; ++kt) {
        int buf = kt & 1;
        if (kt + 1 < KT) stage(buf ^ 1, (kt + 1) * OBK);
        v16bf af[2], bfr[2];
#pragma unroll
        for (int fm = 0; fm < 2; ++fm)
            af[fm] = ldfragA((const bf16_t*)&sA[buf][(wm + fm * 16 + mr) * OBK], half);
#pragma unroll
        for (int fn = 0; fn < 2; ++fn)
            bfr[fn] = ldfragB((const bf16_t*)&sB[buf][(wn + fn * 16 + mr) * OBK], half);
#pragma unroll
        for (int fm = 0; fm < 2; ++fm)
#pragma unroll
            for (int fn = 0; fn < 2; ++fn)
                acc[fm][fn] = __builtin_amdgcn_wmma_f32_16x16x32_bf16(
                    false, af[fm], false, bfr[fn], (short)0, acc[fm][fn],
                    false, false);
        __syncthreads();
    }

#pragma unroll
    for (int fm = 0; fm < 2; ++fm) {
#pragma unroll
        for (int fn = 0; fn < 2; ++fn) {
            int col = bn + wn + fn * 16 + mr;
            float bv = bias[col];
#pragma unroll
            for (int r = 0; r < 8; ++r) {
                int grow = bm + wm + fm * 16 + half * 8 + r;   // = t*B + b
                int bb = grow & (BSZ - 1);
                int tt = grow >> 5;
                out[((size_t)bb * TDEC + tt) * (size_t)VOCAB + col] =
                    acc[fm][fn][r] + bv;
            }
        }
    }
}

// ---------------------------------------------------------------------------
extern "C" void kernel_launch(void* const* d_in, const int* in_sizes, int n_in,
                              void* d_out, int out_size, void* d_ws, size_t ws_size,
                              hipStream_t stream) {
    const int*   X     = (const int*)d_in[0];
    const float* enc   = (const float*)d_in[1];
    const float* hid   = (const float*)d_in[2];
    const int*   vlens = (const int*)d_in[3];
    const float* emb   = (const float*)d_in[4];
    const float* Wq    = (const float*)d_in[5];
    const float* Wk    = (const float*)d_in[6];
    const float* wv    = (const float*)d_in[7];
    const float* Wih0  = (const float*)d_in[8];
    const float* Whh0  = (const float*)d_in[9];
    const float* bih0  = (const float*)d_in[10];
    const float* bhh0  = (const float*)d_in[11];
    const float* Wih1  = (const float*)d_in[12];
    const float* Whh1  = (const float*)d_in[13];
    const float* bih1  = (const float*)d_in[14];
    const float* bhh1  = (const float*)d_in[15];
    const float* Wout  = (const float*)d_in[16];
    const float* bout  = (const float*)d_in[17];
    float* out = (float*)d_out;

    // workspace carve-out (~131 MB), 256B aligned slices
    char* ws = (char*)d_ws;
    size_t off = 0;
    auto alloc = [&](size_t bytes) -> char* {
        char* p = ws + off;
        off = (off + bytes + 255) & ~(size_t)255;
        return p;
    };
    unsigned short* enc_b  = (unsigned short*)alloc((size_t)BSZ * SENC * HDIM * 2);
    unsigned short* Wq_b   = (unsigned short*)alloc((size_t)HDIM * HDIM * 2);
    unsigned short* Wk_b   = (unsigned short*)alloc((size_t)HDIM * HDIM * 2);
    unsigned short* Wih0_b = (unsigned short*)alloc((size_t)G3 * HE * 2);
    unsigned short* Whh0_b = (unsigned short*)alloc((size_t)G3 * HDIM * 2);
    unsigned short* Wih1_b = (unsigned short*)alloc((size_t)G3 * HDIM * 2);
    unsigned short* Whh1_b = (unsigned short*)alloc((size_t)G3 * HDIM * 2);
    unsigned short* Wout_b = (unsigned short*)alloc((size_t)VOCAB * HDIM * 2);
    unsigned short* xs_b   = (unsigned short*)alloc((size_t)TDEC * BSZ * EDIM * 2);
    float* keys  = (float*)alloc((size_t)BSZ * SENC * HDIM * 4);
    float* qbuf  = (float*)alloc((size_t)BSZ * HDIM * 4);
    float* ctx   = (float*)alloc((size_t)BSZ * HDIM * 4);
    unsigned short* inp_b  = (unsigned short*)alloc((size_t)BSZ * HE * 2);
    float* gi    = (float*)alloc((size_t)BSZ * G3 * 4);
    float* gh    = (float*)alloc((size_t)BSZ * G3 * 4);
    float* h0    = (float*)alloc((size_t)BSZ * HDIM * 4);
    float* h1    = (float*)alloc((size_t)BSZ * HDIM * 4);
    unsigned short* h0b = (unsigned short*)alloc((size_t)BSZ * HDIM * 2);
    unsigned short* h1b = (unsigned short*)alloc((size_t)BSZ * HDIM * 2);
    unsigned short* outs_b = (unsigned short*)alloc((size_t)TDEC * BSZ * HDIM * 2);

    auto cvt = [&](const float* s, unsigned short* d, size_t n) {
        s2s_cvt_bf16<<<(unsigned)((n + 255) / 256), 256, 0, stream>>>(s, d, (int)n);
    };
    auto gemm = [&](const unsigned short* A, const unsigned short* W,
                    const float* bias, float* C, int M, int N, int K) {
        dim3 g((unsigned)(N / 64), (unsigned)(M / 32));
        s2s_gemm_bt<<<g, 256, 0, stream>>>(A, W, bias, C, M, N, K);
    };

    // one-time bf16 conversions
    cvt(enc,  enc_b,  (size_t)BSZ * SENC * HDIM);
    cvt(Wq,   Wq_b,   (size_t)HDIM * HDIM);
    cvt(Wk,   Wk_b,   (size_t)HDIM * HDIM);
    cvt(Wih0, Wih0_b, (size_t)G3 * HE);
    cvt(Whh0, Whh0_b, (size_t)G3 * HDIM);
    cvt(Wih1, Wih1_b, (size_t)G3 * HDIM);
    cvt(Whh1, Whh1_b, (size_t)G3 * HDIM);
    cvt(Wout, Wout_b, (size_t)VOCAB * HDIM);

    s2s_embed<<<(TDEC * BSZ * EDIM) / 256, 256, 0, stream>>>(X, emb, xs_b);
    s2s_init_hidden<<<(BSZ * HDIM) / 256, 256, 0, stream>>>(hid, h0, h1, h0b, h1b);

    // keys_proj = enc @ Wk^T   [B*S, H]
    gemm(enc_b, Wk_b, nullptr, keys, BSZ * SENC, HDIM, HDIM);

    // sequential decode steps
    for (int t = 0; t < TDEC; ++t) {
        gemm(h1b, Wq_b, nullptr, qbuf, BSZ, HDIM, HDIM);
        s2s_attn<<<BSZ, 256, 0, stream>>>(qbuf, keys, enc, wv, vlens, ctx);
        s2s_pack_inp<<<(BSZ * HE) / 256, 256, 0, stream>>>(
            ctx, xs_b + (size_t)t * BSZ * EDIM, inp_b);
        // layer 0
        gemm(inp_b, Wih0_b, bih0, gi, BSZ, G3, HE);
        gemm(h0b,   Whh0_b, bhh0, gh, BSZ, G3, HDIM);
        s2s_gru_update<<<(BSZ * HDIM) / 256, 256, 0, stream>>>(
            gi, gh, h0, h0, h0b, nullptr);
        // layer 1
        gemm(h0b, Wih1_b, bih1, gi, BSZ, G3, HDIM);
        gemm(h1b, Whh1_b, bhh1, gh, BSZ, G3, HDIM);
        s2s_gru_update<<<(BSZ * HDIM) / 256, 256, 0, stream>>>(
            gi, gh, h1, h1, h1b, outs_b + (size_t)t * BSZ * HDIM);
    }

    // logits = outs @ W_out^T + b_out, stored as [B, T, V]
    dim3 og(VOCAB / OBN, (TDEC * BSZ) / OBM);
    s2s_out_gemm<<<og, 256, 0, stream>>>(outs_b, Wout_b, bout, out, HDIM);
}
